// TrajectoryPredictionSingle_7687991459871
// MI455X (gfx1250) — matrix operand order repaired and checked by reading, not compiled, not run on hardware
//
#include <hip/hip_runtime.h>
#include <hip/hip_bf16.h>

// ---------------------------------------------------------------------------
// Trajectory prediction pipeline for MI455X (gfx1250, wave32, WMMA + TDM).
//
// Backbone convs = implicit GEMM with v_wmma_f32_16x16x32_f16.
//   * activations NHWC f16; K' = (kh*3+kw)*CIN + cin so each 32-wide K chunk
//     is one kernel position with contiguous channels
//   * weights packed N-major [COUT][9*CIN] f16
//   * per block: the 32-output-channel B panel (32 x 9*CIN f16) is staged
//     into LDS ONCE via the Tensor Data Mover (tensor_load_to_lds +
//     s_wait_tensorcnt), then B fragments are ds_load_b128; A fragments are
//     2 x global b128 loads per lane.  2 N-tiles per wave share one A frag.
// conv1 (CIN=3, 7x7) uses a generic gather path and emits NHWC f16.
// conv4 emits NHWC f32 for the ROI pooling path.
// ---------------------------------------------------------------------------

typedef __attribute__((ext_vector_type(16))) _Float16 v16h;
typedef __attribute__((ext_vector_type(8)))  _Float16 v8h;
typedef __attribute__((ext_vector_type(8)))  float    v8f;
typedef __attribute__((ext_vector_type(4)))  unsigned int v4u;
typedef __attribute__((ext_vector_type(8)))  int      v8i;
typedef __attribute__((ext_vector_type(4)))  int      v4i;

#define BB   8
#define NN   64
#define HIST 7
#define PRED 10

// ---------------------------------------------------------------------------
// Pack OIHW f32 weights -> K-major f16 [KPAD][O]  (conv1 generic path).
// ---------------------------------------------------------------------------
__global__ __launch_bounds__(256) void pack_weights_kmajor_kernel(
    const float* __restrict__ w, _Float16* __restrict__ out,
    int O, int I, int KH, int KW, int KPAD)
{
    int idx = blockIdx.x * 256 + threadIdx.x;
    int total = O * KPAD;
    if (idx >= total) return;
    int n = idx % O;
    int k = idx / O;
    int kreal = I * KH * KW;
    _Float16 v = (_Float16)0.0f;
    if (k < kreal) {
        int cin = k / (KH * KW);
        int r   = k % (KH * KW);
        int kh  = r / KW;
        int kw  = r % KW;
        v = (_Float16)w[(((size_t)n * I + cin) * KH + kh) * KW + kw];
    }
    out[(size_t)k * O + n] = v;
}

// ---------------------------------------------------------------------------
// Pack OIHW f32 3x3 weights -> N-major f16 [O][9*CIN], K' = pos*CIN + cin.
// ---------------------------------------------------------------------------
__global__ __launch_bounds__(256) void pack_weights_nmajor_kernel(
    const float* __restrict__ w, _Float16* __restrict__ out, int O, int I)
{
    int idx = blockIdx.x * 256 + threadIdx.x;
    int ktot = 9 * I;
    int total = O * ktot;
    if (idx >= total) return;
    int o = idx / ktot;
    int r = idx % ktot;
    int p = r / I;
    int cin = r % I;
    int kh = p / 3, kw = p % 3;
    out[idx] = (_Float16)w[(((size_t)o * I + cin) * 3 + kh) * 3 + kw];
}

// ---------------------------------------------------------------------------
// conv1: generic implicit-GEMM WMMA (scattered A gather, NCHW f32 input),
// output NHWC f16.  grid = (tilesM/8, COUT/16, images), block = 256.
// ---------------------------------------------------------------------------
template<int CIN, int COUT, int HIN, int WIN, int HOUT, int WOUT,
         int KH, int KW, int STR, int PADDING, int KPAD>
__global__ __launch_bounds__(256) void conv_wmma_generic_kernel(
    const float* __restrict__ act_in, const _Float16* __restrict__ wgt,
    const float* __restrict__ bias, _Float16* __restrict__ act_out)
{
    constexpr int KREAL = CIN * KH * KW;
    const int lane = threadIdx.x & 31;
    const int wave = threadIdx.x >> 5;
    const int half = lane >> 4;
    const int ln15 = lane & 15;

    const int mtile = blockIdx.x * 8 + wave;
    const int m0 = mtile * 16;
    const int n0 = blockIdx.y * 16;
    const int img = blockIdx.z;

    const int mrow = m0 + ln15;
    const int oh = mrow / WOUT;
    const int ow = mrow % WOUT;

    v8f acc = {};

    for (int k0 = 0; k0 < KPAD; k0 += 32) {
        if (k0 + 32 < KPAD) {
            __builtin_prefetch((const void*)(wgt + (size_t)(k0 + 32) * COUT + n0), 0, 3);
        }
        v16h a;
#pragma unroll
        for (int j = 0; j < 16; ++j) {
            int k = k0 + (half << 4) + j;
            _Float16 v = (_Float16)0.0f;
            if (k < KREAL) {
                int cin = k / (KH * KW);
                int r   = k % (KH * KW);
                int kh  = r / KW;
                int kw  = r % KW;
                int hi = oh * STR - PADDING + kh;
                int wi = ow * STR - PADDING + kw;
                if (hi >= 0 && hi < HIN && wi >= 0 && wi < WIN) {
                    v = (_Float16)act_in[(((size_t)img * CIN + cin) * HIN + hi) * WIN + wi];
                }
            }
            a[j] = v;
        }
        v16h b;
        {
            const _Float16* wp = wgt + (size_t)(k0 + (half << 4)) * COUT + n0 + ln15;
#pragma unroll
            for (int j = 0; j < 16; ++j) b[j] = wp[(size_t)j * COUT];
        }
        acc = __builtin_amdgcn_wmma_f32_16x16x32_f16(
            false, a, false, b, (short)0, acc, false, false);
    }

    const int n = n0 + ln15;
    const float bv = bias[n];
#pragma unroll
    for (int r = 0; r < 8; ++r) {
        int m = m0 + (half << 3) + r;
        int oh2 = m / WOUT;
        int ow2 = m % WOUT;
        float v = acc[r] + bv;
        v = v > 0.0f ? v : 0.0f;
        act_out[(((size_t)img * HOUT + oh2) * WOUT + ow2) * COUT + n] = (_Float16)v;
    }
}

// ---------------------------------------------------------------------------
// Fast 3x3 stride-2 pad-1 conv: NHWC f16 in, WMMA, NHWC out (f16 or f32).
// B panel (32 x 9*CIN f16) staged in LDS via Tensor Data Mover.
// grid = (tilesM/8, COUT/32, images), block = 256,
// dynamic LDS = 32 * 9*CIN * 2 bytes.
// ---------------------------------------------------------------------------
template<int CIN, int COUT, int HIN, int WIN, int HOUT, int WOUT, bool OF32>
__global__ __launch_bounds__(256) void conv3x3_wmma_nhwc_kernel(
    const _Float16* __restrict__ act_in, const _Float16* __restrict__ wgtT,
    const float* __restrict__ bias, void* __restrict__ act_out)
{
    constexpr int KTOT = 9 * CIN;
    constexpr unsigned PANEL_F16 = 32u * (unsigned)KTOT;   // f16 elements
    constexpr unsigned NU = PANEL_F16 / 4u;                // 8-byte units

    extern __shared__ _Float16 bsmem[];                    // [32][KTOT]

    const int lane = threadIdx.x & 31;
    const int wave = threadIdx.x >> 5;
    const int half = lane >> 4;
    const int ln15 = lane & 15;

    const int mtile = blockIdx.x * 8 + wave;
    const int m0 = mtile * 16;
    const int n0 = blockIdx.y * 32;          // two 16-wide N tiles
    const int img = blockIdx.z;

    // ---- stage the B panel (rows n0..n0+31 of wgtT, contiguous) into LDS
#if __has_builtin(__builtin_amdgcn_tensor_load_to_lds) && \
    __has_builtin(__builtin_amdgcn_s_wait_tensorcnt)
    if (threadIdx.x < 32) {     // wave 0 issues one TDM descriptor
        unsigned long long ga =
            (unsigned long long)(uintptr_t)(wgtT + (size_t)n0 * KTOT);
        unsigned lds_off = (unsigned)(uintptr_t)bsmem;
        v4u g0;
        g0[0] = 1u;                                        // count=1 (user D#)
        g0[1] = lds_off;                                   // lds_addr (bytes)
        g0[2] = (unsigned)(ga & 0xffffffffu);              // global_addr[31:0]
        g0[3] = (unsigned)((ga >> 32) & 0x01ffffffu) | (2u << 30);  // type=2
        v8i g1;
        g1[0] = (int)(3u << 16);                           // data_size = 8B
        g1[1] = (int)((NU & 0xffffu) << 16);               // tensor_dim0[15:0]
        g1[2] = (int)((NU >> 16) | (1u << 16));            // dim0[31:16], dim1=1
        g1[3] = (int)((NU & 0xffffu) << 16);               // tile_dim0 = NU
        g1[4] = 1;                                         // tile_dim1 = 1
        g1[5] = (int)NU;                                   // dim0_stride
        g1[6] = 0;
        g1[7] = 0;
        v4i gz = {0, 0, 0, 0};
#if defined(__clang_major__) && (__clang_major__ >= 23)
        v8i gz8 = {0, 0, 0, 0, 0, 0, 0, 0};
        __builtin_amdgcn_tensor_load_to_lds(g0, g1, gz, gz, gz8, 0);
#else
        __builtin_amdgcn_tensor_load_to_lds(g0, g1, gz, gz, 0);
#endif
        __builtin_amdgcn_s_wait_tensorcnt(0);
    }
#else
    // fallback: cooperative copy (global b128 loads + ds stores)
    {
        const v8h* src = (const v8h*)(wgtT + (size_t)n0 * KTOT);
        v8h* dst = (v8h*)bsmem;
        for (unsigned i = threadIdx.x; i < PANEL_F16 / 8u; i += 256u)
            dst[i] = src[i];
    }
#endif
    __syncthreads();

    const int mrow = m0 + ln15;
    const int oh = mrow / WOUT;
    const int ow = mrow % WOUT;

    v8f acc0 = {}, acc1 = {};
    const v8h vzero = {};

    // per-lane B row pointers in LDS (include the half*16 K offset)
    const _Float16* brow0 = bsmem + (size_t)ln15 * KTOT + (half << 4);
    const _Float16* brow1 = brow0 + (size_t)16 * KTOT;

    for (int p = 0; p < 9; ++p) {
        const int kh = p / 3, kw = p % 3;
        int hi = oh * 2 - 1 + kh;
        int wi = ow * 2 - 1 + kw;
        const bool valid = (hi >= 0) && (hi < HIN) && (wi >= 0) && (wi < WIN);
        hi = min(max(hi, 0), HIN - 1);
        wi = min(max(wi, 0), WIN - 1);
        const _Float16* ap = act_in + (((size_t)img * HIN + hi) * WIN + wi) * CIN + (half << 4);
        const _Float16* bp0 = brow0 + p * CIN;
        const _Float16* bp1 = brow1 + p * CIN;
#pragma unroll
        for (int c0 = 0; c0 < CIN; c0 += 32) {
            // ---- A fragment: channels c0+half*16.. and c0+half*16+8..
            v8h alo = *(const v8h*)(ap + c0);
            v8h ahi = *(const v8h*)(ap + c0 + 8);
            if (!valid) { alo = vzero; ahi = vzero; }
            v16h a;
#pragma unroll
            for (int j = 0; j < 8; ++j) { a[j] = alo[j]; a[j + 8] = ahi[j]; }
            // ---- B fragments from LDS: 16 contiguous f16 each
            v8h b0lo = *(const v8h*)(bp0 + c0);
            v8h b0hi = *(const v8h*)(bp0 + c0 + 8);
            v8h b1lo = *(const v8h*)(bp1 + c0);
            v8h b1hi = *(const v8h*)(bp1 + c0 + 8);
            v16h b0, b1;
#pragma unroll
            for (int j = 0; j < 8; ++j) {
                b0[j] = b0lo[j]; b0[j + 8] = b0hi[j];
                b1[j] = b1lo[j]; b1[j + 8] = b1hi[j];
            }
            acc0 = __builtin_amdgcn_wmma_f32_16x16x32_f16(
                false, a, false, b0, (short)0, acc0, false, false);
            acc1 = __builtin_amdgcn_wmma_f32_16x16x32_f16(
                false, a, false, b1, (short)0, acc1, false, false);
        }
    }

    const int na = n0 + ln15;
    const int nb = na + 16;
    const float bva = bias[na];
    const float bvb = bias[nb];
#pragma unroll
    for (int r = 0; r < 8; ++r) {
        int m = m0 + (half << 3) + r;
        int oh2 = m / WOUT;
        int ow2 = m % WOUT;
        size_t pix = (((size_t)img * HOUT + oh2) * WOUT + ow2) * COUT;
        float v0 = acc0[r] + bva; v0 = v0 > 0.0f ? v0 : 0.0f;
        float v1 = acc1[r] + bvb; v1 = v1 > 0.0f ? v1 : 0.0f;
        if (OF32) {
            ((float*)act_out)[pix + na] = v0;
            ((float*)act_out)[pix + nb] = v1;
        } else {
            ((_Float16*)act_out)[pix + na] = (_Float16)v0;
            ((_Float16*)act_out)[pix + nb] = (_Float16)v1;
        }
    }
}

// ---------------------------------------------------------------------------
// ROI mean pooling over NHWC f32 feature map (B,32,32,512).
// grid = B*N, block = 256 (lane-coalesced over channels).
// ---------------------------------------------------------------------------
__global__ __launch_bounds__(256) void roi_pool_kernel(
    const float* __restrict__ fm,    // (B,32,32,512) NHWC f32
    const float* __restrict__ rois,  // (B,64,4,2)
    const float* __restrict__ sf,    // (B,)
    float* __restrict__ pooled)      // (B*64, 512)
{
    int bn = blockIdx.x;
    int b = bn >> 6;
    float s = (32.0f / 512.0f) * sf[b];
    const float* r = rois + (size_t)bn * 8;
    float xmn = r[0], xmx = r[0], ymn = r[1], ymx = r[1];
#pragma unroll
    for (int c = 1; c < 4; ++c) {
        xmn = fminf(xmn, r[c * 2]);     xmx = fmaxf(xmx, r[c * 2]);
        ymn = fminf(ymn, r[c * 2 + 1]); ymx = fmaxf(ymx, r[c * 2 + 1]);
    }
    int xmin = min(max((int)truncf(xmn * s), 0), 31);
    int xmax = min(max((int)truncf(xmx * s), 0), 31);
    int ymin = min(max((int)truncf(ymn * s), 0), 31);
    int ymax = min(max((int)truncf(ymx * s), 0), 31);
    float inv_cnt = 1.0f / (float)((xmax - xmin + 1) * (ymax - ymin + 1));
    for (int c = threadIdx.x; c < 512; c += 256) {
        float acc = 0.0f;
        for (int x = xmin; x <= xmax; ++x)
            for (int y = ymin; y <= ymax; ++y)
                acc += fm[(((size_t)b * 32 + x) * 32 + y) * 512 + c];
        pooled[(size_t)bn * 512 + c] = acc * inv_cnt;
    }
}

// ---------------------------------------------------------------------------
// roi_feat = relu(pooled @ pool_w.T + pool_b).  grid = B*N, block = 64.
// ---------------------------------------------------------------------------
__global__ __launch_bounds__(64) void roi_fc_kernel(
    const float* __restrict__ pooled, const float* __restrict__ pw,
    const float* __restrict__ pb, float* __restrict__ roi_feat)
{
    int bn = blockIdx.x;
    int t = threadIdx.x;
    if (t < 50) {
        float acc = pb[t];
        const float* p = pooled + (size_t)bn * 512;
        const float* w = pw + (size_t)t * 512;
        for (int i = 0; i < 512; ++i) acc += p[i] * w[i];
        roi_feat[(size_t)bn * 50 + t] = acc > 0.0f ? acc : 0.0f;
    }
}

// ---------------------------------------------------------------------------
// Displacement embedding: conv1d(1->64,k3,VALID)+relu, flatten, 320->50+relu.
// grid = (B*N, axis), block = 64.
// ---------------------------------------------------------------------------
__global__ __launch_bounds__(64) void disp_embed_kernel(
    const float* __restrict__ obs,   // (B,64,7,2)
    const float* __restrict__ cwx, const float* __restrict__ cbx,
    const float* __restrict__ lwx, const float* __restrict__ lbx,
    const float* __restrict__ cwy, const float* __restrict__ cby,
    const float* __restrict__ lwy, const float* __restrict__ lby,
    float* __restrict__ pred)        // (B*64, 2, 50)
{
    int bn = blockIdx.x;
    int axis = blockIdx.y;
    const float* cw = axis ? cwy : cwx;
    const float* cb = axis ? cby : cbx;
    const float* lw = axis ? lwy : lwx;
    const float* lb = axis ? lby : lbx;

    __shared__ float seq[HIST];
    __shared__ float feat[320];
    int t = threadIdx.x;
    if (t < HIST) seq[t] = obs[(size_t)bn * (HIST * 2) + t * 2 + axis];
    __syncthreads();
    {
        int c = t;   // t in [0,64)
        float w0 = cw[c * 3 + 0], w1 = cw[c * 3 + 1], w2 = cw[c * 3 + 2];
        float bb = cb[c];
#pragma unroll
        for (int p = 0; p < 5; ++p) {
            float v = bb + seq[p] * w0 + seq[p + 1] * w1 + seq[p + 2] * w2;
            feat[c * 5 + p] = v > 0.0f ? v : 0.0f;
        }
    }
    __syncthreads();
    if (t < 50) {
        float acc = lb[t];
        const float* w = lw + (size_t)t * 320;
        for (int i = 0; i < 320; ++i) acc += feat[i] * w[i];
        acc = acc > 0.0f ? acc : 0.0f;
        pred[((size_t)bn * 2 + axis) * 50 + t] = acc;
    }
}

// ---------------------------------------------------------------------------
// Heads: [pred_axis(50) | roi_feat(50)] @ p{x,y}_w.T + bias -> out.
// grid = B*N, block = 32.
// ---------------------------------------------------------------------------
__global__ __launch_bounds__(32) void head_kernel(
    const float* __restrict__ pred, const float* __restrict__ roi_feat,
    const float* __restrict__ pxw, const float* __restrict__ pxb,
    const float* __restrict__ pyw, const float* __restrict__ pyb,
    float* __restrict__ out)         // (B*N, PRED, 2)
{
    int bn = blockIdx.x;
    int t = threadIdx.x;
    if (t < 20) {
        int axis = t / 10, p = t % 10;
        const float* w  = axis ? pyw : pxw;
        const float* bv = axis ? pyb : pxb;
        const float* e  = pred + ((size_t)bn * 2 + axis) * 50;
        const float* rf = roi_feat + (size_t)bn * 50;
        float acc = bv[p];
        const float* wp = w + (size_t)p * 100;
        for (int i = 0; i < 50; ++i) acc += e[i] * wp[i];
        for (int i = 0; i < 50; ++i) acc += rf[i] * wp[50 + i];
        out[((size_t)bn * PRED + p) * 2 + axis] = acc;
    }
}

// ---------------------------------------------------------------------------
extern "C" void kernel_launch(void* const* d_in, const int* in_sizes, int n_in,
                              void* d_out, int out_size, void* d_ws, size_t ws_size,
                              hipStream_t stream) {
    (void)in_sizes; (void)n_in; (void)out_size; (void)ws_size;

    const float* imgs = (const float*)d_in[0];
    const float* obs  = (const float*)d_in[1];
    const float* rois = (const float*)d_in[2];
    const float* sf   = (const float*)d_in[3];
    const float* c1xw = (const float*)d_in[4];
    const float* c1xb = (const float*)d_in[5];
    const float* lxw  = (const float*)d_in[6];
    const float* lxb  = (const float*)d_in[7];
    const float* c1yw = (const float*)d_in[8];
    const float* c1yb = (const float*)d_in[9];
    const float* lyw  = (const float*)d_in[10];
    const float* lyb  = (const float*)d_in[11];
    const float* bbw1 = (const float*)d_in[12];
    const float* bbb1 = (const float*)d_in[13];
    const float* bbw2 = (const float*)d_in[14];
    const float* bbb2 = (const float*)d_in[15];
    const float* bbw3 = (const float*)d_in[16];
    const float* bbb3 = (const float*)d_in[17];
    const float* bbw4 = (const float*)d_in[18];
    const float* bbb4 = (const float*)d_in[19];
    const float* poolw = (const float*)d_in[20];
    const float* poolb = (const float*)d_in[21];
    const float* pxw  = (const float*)d_in[22];
    const float* pxb  = (const float*)d_in[23];
    const float* pyw  = (const float*)d_in[24];
    const float* pyb  = (const float*)d_in[25];
    float* out = (float*)d_out;

    // --- workspace carve-up (256B aligned) ---
    char* ws = (char*)d_ws;
    size_t off = 0;
    auto carve = [&](size_t bytes) -> void* {
        off = (off + 255) & ~(size_t)255;
        void* p = ws + off;
        off += bytes;
        return p;
    };
    constexpr int KP1 = 160;   // conv1: 3*7*7 = 147 -> pad to 160

    _Float16* w1h = (_Float16*)carve((size_t)KP1 * 64 * 2);          // [KP1][64]
    _Float16* w2h = (_Float16*)carve((size_t)128 * 9 * 64  * 2);     // [128][576]
    _Float16* w3h = (_Float16*)carve((size_t)256 * 9 * 128 * 2);     // [256][1152]
    _Float16* w4h = (_Float16*)carve((size_t)512 * 9 * 256 * 2);     // [512][2304]

    size_t x1_elems = (size_t)BB * 256 * 256 * 64;    // NHWC f16
    size_t x2_elems = (size_t)BB * 128 * 128 * 128;   // NHWC f16
    _Float16* x1 = (_Float16*)carve(x1_elems * 2);    // reused for x3
    _Float16* x2 = (_Float16*)carve(x2_elems * 2);    // reused for fm (f32)
    _Float16* x3 = x1;                                // 8*64*64*256 f16 = 16MB < 64MB
    float*    fm = (float*)x2;                        // 8*32*32*512 f32 = 16MB < 32MB

    float* pooled   = (float*)carve((size_t)BB * NN * 512 * 4);
    float* roi_feat = (float*)carve((size_t)BB * NN * 50 * 4);
    float* pred     = (float*)carve((size_t)BB * NN * 2 * 50 * 4);

    auto packgrid = [](int total) { return (total + 255) / 256; };
    pack_weights_kmajor_kernel<<<packgrid(64 * KP1), 256, 0, stream>>>(
        bbw1, w1h, 64, 3, 7, 7, KP1);
    pack_weights_nmajor_kernel<<<packgrid(128 * 9 * 64), 256, 0, stream>>>(bbw2, w2h, 128, 64);
    pack_weights_nmajor_kernel<<<packgrid(256 * 9 * 128), 256, 0, stream>>>(bbw3, w3h, 256, 128);
    pack_weights_nmajor_kernel<<<packgrid(512 * 9 * 256), 256, 0, stream>>>(bbw4, w4h, 512, 256);

    // --- backbone ---
    // conv1: 3->64, 512->256, 7x7 s2 p3, generic path, out NHWC f16.
    conv_wmma_generic_kernel<3, 64, 512, 512, 256, 256, 7, 7, 2, 3, KP1>
        <<<dim3(4096 / 8, 64 / 16, BB), 256, 0, stream>>>(imgs, w1h, bbb1, x1);
    // conv2: 64->128, 256->128.  LDS panel = 32*576*2 = 36864 B
    conv3x3_wmma_nhwc_kernel<64, 128, 256, 256, 128, 128, false>
        <<<dim3(1024 / 8, 128 / 32, BB), 256, 36864, stream>>>(x1, w2h, bbb2, x2);
    // conv3: 128->256, 128->64.  LDS panel = 32*1152*2 = 73728 B
    conv3x3_wmma_nhwc_kernel<128, 256, 128, 128, 64, 64, false>
        <<<dim3(256 / 8, 256 / 32, BB), 256, 73728, stream>>>(x2, w3h, bbb3, x3);
    // conv4: 256->512, 64->32, out NHWC f32.  LDS panel = 32*2304*2 = 147456 B
    conv3x3_wmma_nhwc_kernel<256, 512, 64, 64, 32, 32, true>
        <<<dim3(64 / 8, 512 / 32, BB), 256, 147456, stream>>>(x3, w4h, bbb4, fm);

    // --- ROI pooling + FC ---
    roi_pool_kernel<<<BB * NN, 256, 0, stream>>>(fm, rois, sf, pooled);
    roi_fc_kernel<<<BB * NN, 64, 0, stream>>>(pooled, poolw, poolb, roi_feat);

    // --- displacement embeddings ---
    disp_embed_kernel<<<dim3(BB * NN, 2), 64, 0, stream>>>(
        obs, c1xw, c1xb, lxw, lxb, c1yw, c1yb, lyw, lyb, pred);

    // --- heads ---
    head_kernel<<<BB * NN, 32, 0, stream>>>(pred, roi_feat, pxw, pxb, pyw, pyb, out);
}
